// LocalTriangleAttentionNew_20375324852311
// MI455X (gfx1250) — matrix operand order repaired and checked by reading, not compiled, not run on hardware
//
#include <hip/hip_runtime.h>
#include <stdint.h>

// Triangle multiplicative update (outgoing + incoming), N=512, C=128, fp32 I/O,
// bf16 WMMA (v_wmma_f32_16x16x32_bf16) for all GEMMs with f32 accumulation.
//
// d_in ordering assumption (setup_inputs() dict insertion order, nested dicts
// flattened in place):
//   0: edge_embed [1,512,512,128] f32
//   1: edge_mask  [1,512,512]     f32
//   2..17 : params_out leaves in insertion order:
//     +0 ln_in_w +1 ln_in_b +2 a_p_w +3 a_p_b +4 a_g_w +5 a_g_b
//     +6 b_p_w +7 b_p_b +8 b_g_w +9 b_g_b +10 ln_out_w +11 ln_out_b
//     +12 g_w +13 g_b +14 z_w +15 z_b
//   18..33: params_in, same layout.

#define N_DIM 512
#define NN (N_DIM * N_DIM)
#define C_DIM 128

typedef __bf16    v16bf __attribute__((ext_vector_type(16)));
typedef float     v8f   __attribute__((ext_vector_type(8)));
typedef uint32_t  u32x4 __attribute__((ext_vector_type(4)));

union ABPack { v16bf v; u32x4 q[2]; };

struct TriParams {
  const float *ln_in_w, *ln_in_b;
  const float *a_p_b, *a_g_b, *b_p_b, *b_g_b;
  const float *ln_out_w, *ln_out_b, *g_b, *z_b;
  const uint16_t *w_ap, *w_ag, *w_bp, *w_bg, *w_g, *w_z; // bf16 bits, [128][128] row-major [out][in]
};

__device__ __forceinline__ uint16_t bfbits(float f) {
  union { float f; uint32_t u; } v; v.f = f;
  uint32_t u = v.u + 0x7fffu + ((v.u >> 16) & 1u); // round-to-nearest-even
  return (uint16_t)(u >> 16);
}

__device__ __forceinline__ float sigm(float x) { return 1.0f / (1.0f + __expf(-x)); }

// A-matrix (16x32 bf16, M=row per lane): lane = 16*half + m holds
// K = kBase + 8*half + [0..7] and K = kBase + 16 + 8*half + [0..7].  (ISA 7.12.2)
__device__ __forceinline__ v16bf load_mat_a(const uint16_t* base, int stride,
                                            int row, int half, int kBase) {
  ABPack u;
  const uint16_t* p = base + row * stride + kBase + 8 * half;
  u.q[0] = *(const u32x4*)(p);
  u.q[1] = *(const u32x4*)(p + 16);
  return u.v;
}

// B-matrix (32x16 bf16, N=col per lane): lane = 16*half + n holds
// K = kBase + 8*half + [0..7] and K = kBase + 16 + 8*half + [0..7] of column n; B[k][n] == W[n][k] (row-major W).
__device__ __forceinline__ v16bf load_mat_b(const uint16_t* base, int stride,
                                            int col, int half, int kBase) {
  ABPack u;
  const uint16_t* p = base + col * stride + kBase + 8 * half;
  u.q[0] = *(const u32x4*)(p);
  u.q[1] = *(const u32x4*)(p + 16);
  return u.v;
}

__device__ __forceinline__ v8f wmma_bf16(v16bf a, v16bf b, v8f c) {
  return __builtin_amdgcn_wmma_f32_16x16x32_bf16(false, a, false, b, (short)0, c,
                                                 false, false);
}

// LayerNorm of one 128-ch row (2 lanes per row: half=0 -> ch 0..63, half=1 -> 64..127),
// writes bf16 result to LDS row.
__device__ __forceinline__ void ln_row_to_lds(const float* __restrict__ src_row,
                                              const float* __restrict__ gw,
                                              const float* __restrict__ gb,
                                              uint16_t* __restrict__ dst_row,
                                              int half) {
  const float* p = src_row + half * 64;
  float vals[64];
  float s = 0.f, s2 = 0.f;
#pragma unroll
  for (int c = 0; c < 64; ++c) { float v = p[c]; vals[c] = v; s += v; s2 += v * v; }
  s  += __shfl_xor(s, 16, 32);
  s2 += __shfl_xor(s2, 16, 32);
  const float mean = s * (1.f / 128.f);
  const float rs = rsqrtf(s2 * (1.f / 128.f) - mean * mean + 1e-5f);
#pragma unroll
  for (int c = 0; c < 64; ++c) {
    const int ch = half * 64 + c;
    dst_row[ch] = bfbits((vals[c] - mean) * rs * gw[ch] + gb[ch]);
  }
}

// ---------------- K0: fp32 -> bf16 weight conversion (128x128 matrix) -------
__global__ __launch_bounds__(256) void k_conv(const float* __restrict__ s,
                                              uint16_t* __restrict__ d) {
  const int i = blockIdx.x * 256 + threadIdx.x; // grid 64 -> 16384 exactly
  d[i] = bfbits(s[i]);
}

// ---------------- K1: LN + a/b projections, store channel-major ------------
// a_t/b_t layout: [c][row][col] bf16. Outgoing: value(rhi,rlo,c) -> [c][rhi][rlo].
// Incoming:  value(rhi,rlo,c) -> [c][rlo][rhi]  (einsum 'bkic','bkjc' transpose).
__global__ __launch_bounds__(256) void k_proj(const float* __restrict__ zin,
                                              const float* __restrict__ mask,
                                              TriParams P,
                                              uint16_t* __restrict__ a_t,
                                              uint16_t* __restrict__ b_t,
                                              int incoming) {
  __shared__ __align__(16) uint16_t zl[8][16][C_DIM];
  __shared__ float mrow[8][16];

  const int tid = threadIdx.x;
  const int wave = tid >> 5, lane = tid & 31;
  const int m = lane & 15, half = lane >> 4;
  const int rbase = blockIdx.x * 128 + wave * 16;
  const int i_hi = rbase >> 9;          // constant over the wave's 16 rows
  const int k_lo = rbase & (N_DIM - 1); // 16-aligned

  {
    const int r = rbase + m;
    ln_row_to_lds(zin + (size_t)r * C_DIM, P.ln_in_w, P.ln_in_b, &zl[wave][m][0], half);
    if (half == 0) mrow[wave][m] = mask[r];
  }
  __syncthreads();

  const uint16_t* Az = &zl[wave][0][0];

  for (int pair = 0; pair < 2; ++pair) {
    const uint16_t* Wp = pair ? P.w_bp : P.w_ap;
    const uint16_t* Wg = pair ? P.w_bg : P.w_ag;
    const float* biasP = pair ? P.b_p_b : P.a_p_b;
    const float* biasG = pair ? P.b_g_b : P.a_g_b;
    uint16_t* dst = pair ? b_t : a_t;

    for (int nb = 0; nb < 8; ++nb) {
      const int o = nb * 16 + m; // this lane's output channel (N index)
      v8f accP = {}, accG = {};
#pragma unroll
      for (int kb = 0; kb < 4; ++kb) {
        const int ks = kb * 32;
        v16bf A = load_mat_a(Az, C_DIM, m, half, ks);
        accP = wmma_bf16(A, load_mat_b(Wp, C_DIM, o, half, ks), accP);
        accG = wmma_bf16(A, load_mat_b(Wg, C_DIM, o, half, ks), accG);
      }
      const float bp = biasP[o], bg = biasG[o];
      if (!incoming) {
        // fixed channel o, 8 consecutive cols -> one 16B store
        u32x4 pk;
#pragma unroll
        for (int j = 0; j < 8; ++j) {
          const float v = mrow[wave][j + 8 * half] * sigm(accG[j] + bg) * (accP[j] + bp);
          const uint32_t h = bfbits(v);
          if (j & 1) pk[j >> 1] |= h << 16; else pk[j >> 1] = h;
        }
        uint16_t* d = dst + (size_t)o * NN + (size_t)i_hi * N_DIM + (k_lo + 8 * half);
        *(u32x4*)d = pk;
      } else {
        // transposed scatter: [c][rlo][rhi]
#pragma unroll
        for (int j = 0; j < 8; ++j) {
          const int m2 = j + 8 * half;
          const float v = mrow[wave][m2] * sigm(accG[j] + bg) * (accP[j] + bp);
          dst[(size_t)o * NN + (size_t)(k_lo + m2) * N_DIM + i_hi] = bfbits(v);
        }
      }
    }
  }
}

// ---------------- K2: triangle einsum x[i,j,c] = sum_k A_c[i,k] * B_c[j,k] --
// grid (8, 4, 128): 64(i) x 128(j) tile per block per channel.
// 8 waves in 2(i) x 4(j); each wave: 32x32 tile = 4 accumulators with 2x2
// register reuse (8 b128 loads per 4 WMMAs).
__global__ __launch_bounds__(256) void k_tri(const uint16_t* __restrict__ a_t,
                                             const uint16_t* __restrict__ b_t,
                                             float* __restrict__ x) {
  const int tid = threadIdx.x;
  const int wave = tid >> 5, lane = tid & 31;
  const int n = lane & 15, half = lane >> 4;
  const int c = blockIdx.z;
  const int i0 = blockIdx.x * 64 + (wave & 1) * 32;
  const int j0 = blockIdx.y * 128 + (wave >> 1) * 32;
  const uint16_t* A = a_t + (size_t)c * NN + (size_t)i0 * N_DIM;
  const uint16_t* B = b_t + (size_t)c * NN;

  v8f acc00 = {}, acc01 = {}, acc10 = {}, acc11 = {};
  for (int kb = 0; kb < 16; ++kb) {
    const int ks = kb * 32;
    v16bf Av0 = load_mat_a(A, N_DIM, n, half, ks);
    v16bf Av1 = load_mat_a(A, N_DIM, n + 16, half, ks);
    v16bf Bv0 = load_mat_b(B, N_DIM, j0 + n, half, ks);
    v16bf Bv1 = load_mat_b(B, N_DIM, j0 + 16 + n, half, ks);
    acc00 = wmma_bf16(Av0, Bv0, acc00);
    acc01 = wmma_bf16(Av0, Bv1, acc01);
    acc10 = wmma_bf16(Av1, Bv0, acc10);
    acc11 = wmma_bf16(Av1, Bv1, acc11);
  }
#pragma unroll
  for (int j = 0; j < 8; ++j) {
    const size_t row0 = (size_t)(i0 + j + 8 * half) * (N_DIM * C_DIM);
    const size_t row1 = (size_t)(i0 + 16 + j + 8 * half) * (N_DIM * C_DIM);
    x[row0 + (size_t)(j0 + n) * C_DIM + c] = acc00[j];
    x[row0 + (size_t)(j0 + 16 + n) * C_DIM + c] = acc01[j];
    x[row1 + (size_t)(j0 + n) * C_DIM + c] = acc10[j];
    x[row1 + (size_t)(j0 + 16 + n) * C_DIM + c] = acc11[j];
  }
}

// ---------------- K3: LN(x), LN(z), out = z + (x_ln@z_w^T + z_b) * sigmoid(z_ln@g_w^T + g_b)
__global__ __launch_bounds__(256) void k_out(const float* zin,
                                             const float* __restrict__ x,
                                             TriParams P, float* zout) {
  __shared__ __align__(16) uint16_t xl[8][16][C_DIM];
  __shared__ __align__(16) uint16_t zl[8][16][C_DIM];

  const int tid = threadIdx.x;
  const int wave = tid >> 5, lane = tid & 31;
  const int m = lane & 15, half = lane >> 4;
  const int rbase = blockIdx.x * 128 + wave * 16;
  const int r = rbase + m;

  ln_row_to_lds(x + (size_t)r * C_DIM, P.ln_out_w, P.ln_out_b, &xl[wave][m][0], half);
  ln_row_to_lds(zin + (size_t)r * C_DIM, P.ln_in_w, P.ln_in_b, &zl[wave][m][0], half);
  __syncthreads();

  const uint16_t* Ax = &xl[wave][0][0];
  const uint16_t* Azl = &zl[wave][0][0];

  for (int nb = 0; nb < 8; ++nb) {
    const int o = nb * 16 + m;
    v8f accZ = {}, accG = {};
#pragma unroll
    for (int kb = 0; kb < 4; ++kb) {
      const int ks = kb * 32;
      accZ = wmma_bf16(load_mat_a(Ax, C_DIM, m, half, ks),
                       load_mat_b(P.w_z, C_DIM, o, half, ks), accZ);
      accG = wmma_bf16(load_mat_a(Azl, C_DIM, m, half, ks),
                       load_mat_b(P.w_g, C_DIM, o, half, ks), accG);
    }
    const float zb = P.z_b[o], gb = P.g_b[o];
#pragma unroll
    for (int j = 0; j < 8; ++j) {
      const size_t off = (size_t)(rbase + j + 8 * half) * C_DIM + o;
      zout[off] = zin[off] + (accZ[j] + zb) * sigm(accG[j] + gb);
    }
  }
}

extern "C" void kernel_launch(void* const* d_in, const int* in_sizes, int n_in,
                              void* d_out, int out_size, void* d_ws, size_t ws_size,
                              hipStream_t stream) {
  (void)in_sizes; (void)n_in; (void)out_size; (void)ws_size;
  const float* edge = (const float*)d_in[0];
  const float* mask = (const float*)d_in[1];

  char* ws = (char*)d_ws;
  uint16_t* wbf = (uint16_t*)ws;                               // 12 * 32KB bf16 weights
  uint16_t* a_t = (uint16_t*)(ws + 0x80000);                   // 64 MiB
  uint16_t* b_t = (uint16_t*)(ws + 0x80000 + 0x4000000);       // 64 MiB
  float*    x   = (float*)   (ws + 0x80000 + 0x8000000);       // 128 MiB

  // Convert the 6 weight matrices per pass: [a_p, a_g, b_p, b_g, g_w, z_w]
  static const int woff[6] = {2, 4, 6, 8, 12, 14};
  for (int pass = 0; pass < 2; ++pass) {
    const int base = 2 + pass * 16;
    for (int w = 0; w < 6; ++w) {
      k_conv<<<64, 256, 0, stream>>>((const float*)d_in[base + woff[w]],
                                     wbf + (size_t)(pass * 6 + w) * 16384);
    }
  }

  for (int pass = 0; pass < 2; ++pass) {
    const int pb = 2 + pass * 16;
    TriParams P;
    P.ln_in_w  = (const float*)d_in[pb + 0];
    P.ln_in_b  = (const float*)d_in[pb + 1];
    P.a_p_b    = (const float*)d_in[pb + 3];
    P.a_g_b    = (const float*)d_in[pb + 5];
    P.b_p_b    = (const float*)d_in[pb + 7];
    P.b_g_b    = (const float*)d_in[pb + 9];
    P.ln_out_w = (const float*)d_in[pb + 10];
    P.ln_out_b = (const float*)d_in[pb + 11];
    P.g_b      = (const float*)d_in[pb + 13];
    P.z_b      = (const float*)d_in[pb + 15];
    P.w_ap = wbf + (size_t)(pass * 6 + 0) * 16384;
    P.w_ag = wbf + (size_t)(pass * 6 + 1) * 16384;
    P.w_bp = wbf + (size_t)(pass * 6 + 2) * 16384;
    P.w_bg = wbf + (size_t)(pass * 6 + 3) * 16384;
    P.w_g  = wbf + (size_t)(pass * 6 + 4) * 16384;
    P.w_z  = wbf + (size_t)(pass * 6 + 5) * 16384;

    const float* zin = pass ? (const float*)d_out : edge;
    k_proj<<<2048, 256, 0, stream>>>(zin, mask, P, a_t, b_t, /*incoming=*/pass);
    k_tri<<<dim3(8, 4, 128), 256, 0, stream>>>(a_t, b_t, x);
    k_out<<<2048, 256, 0, stream>>>(zin, x, P, (float*)d_out);
  }
}
